// NexusGraphSAGE_7310034337833
// MI455X (gfx1250) — matrix operand order repaired and checked by reading, not compiled, not run on hardware
//
#include <hip/hip_runtime.h>

typedef __attribute__((ext_vector_type(16))) _Float16 v16h;
typedef __attribute__((ext_vector_type(8)))  _Float16 v8h;
typedef __attribute__((ext_vector_type(8)))  float    v8f;

#define D_IN   128
#define D_HID  128
#define D_HID2 64

// ---------------------------------------------------------------------------
// Vector helpers
// ---------------------------------------------------------------------------
__device__ __forceinline__ v16h cat16(v8h lo, v8h hi) {
  return __builtin_shufflevector(lo, hi, 0, 1, 2, 3, 4, 5, 6, 7,
                                 8, 9, 10, 11, 12, 13, 14, 15);
}

// A fragment from LDS: 16 consecutive K per lane (halves 0..7 at p, 8..15 at
// p+8), matching the B fragment order. Two ds_load_b128 per fragment.
__device__ __forceinline__ v16h load_a_lds(const _Float16* p) {
  v8h lo = *(const v8h*)p;
  v8h hi = *(const v8h*)(p + 8);
  return cat16(lo, hi);
}

// B fragment from pre-converted f16 weights: 16 consecutive K per lane
// (row-major W[out][in] is B^T). Two global_load_b128 per fragment.
__device__ __forceinline__ v16h load_b_f16(const _Float16* p) {
  v8h lo = *(const v8h*)p;
  v8h hi = *(const v8h*)(p + 8);
  return cat16(lo, hi);
}

// ---------------------------------------------------------------------------
// Fused SAGE layer GEMM: out = relu( (agg*invdeg) @ Wl^T + root @ Wr^T + b ).
// The 16-row A panel (both matrices) is staged in LDS as f16 ONCE per block
// (scale folded in), so the FOUT/16 waves share it instead of each re-loading
// and re-converting it. Rows padded by 8 halves (16B) to de-conflict the
// 16 lanes that read the same column offset of different rows.
// ---------------------------------------------------------------------------
template <int K, int FOUT>
__global__ void sage_gemm_kernel(const float* __restrict__ agg,
                                 const float* __restrict__ invdeg,
                                 const float* __restrict__ root,
                                 const _Float16* __restrict__ Wl,
                                 const _Float16* __restrict__ Wr,
                                 const float* __restrict__ bias,
                                 float* __restrict__ out) {
  constexpr int NTHR  = (FOUT / 16) * 32;  // one wave per 16-col tile
  constexpr int CH    = K / 4;             // float4 chunks per row
  constexpr int PANEL = 16 * CH;           // chunks per panel
  constexpr int LDK   = K + 8;             // padded LDS row (halves)

  __shared__ __align__(16) _Float16 As[2][16][LDK];

  const int m0 = blockIdx.x << 4;

  // ---- cooperative stage: global f32 -> LDS f16 (scaled) ----
  for (int idx = threadIdx.x; idx < 2 * PANEL; idx += NTHR) {
    const int p   = idx / PANEL;
    const int rem = idx - p * PANEL;
    const int row = rem / CH;
    const int c   = rem - row * CH;
    const float* sp = (p ? root : agg) + (size_t)(m0 + row) * K + (c << 2);
    const float4 v = *(const float4*)sp;
    const float  s = p ? 1.0f : invdeg[m0 + row];
    _Float16* dp = &As[p][row][c << 2];
    dp[0] = (_Float16)(v.x * s);
    dp[1] = (_Float16)(v.y * s);
    dp[2] = (_Float16)(v.z * s);
    dp[3] = (_Float16)(v.w * s);
  }
  __syncthreads();

  // ---- WMMA: 16 nodes x 16 features per wave ----
  const int lane = threadIdx.x & 31;
  const int wave = threadIdx.x >> 5;
  const int n0   = wave << 4;
  const int n    = n0 + (lane & 15);

  const _Float16* a0 = &As[0][lane & 15][(lane >> 4) << 4];
  const _Float16* a1 = &As[1][lane & 15][(lane >> 4) << 4];
  const _Float16* bl = Wl + (size_t)n * K + ((lane >> 4) << 4);
  const _Float16* br = Wr + (size_t)n * K + ((lane >> 4) << 4);

  v8f c = {};
#pragma unroll
  for (int kb = 0; kb < K; kb += 32) {     // mean-aggregated neighbors @ Wl^T
    v16h a = load_a_lds(a0 + kb);
    v16h b = load_b_f16(bl + kb);
    c = __builtin_amdgcn_wmma_f32_16x16x32_f16(false, a, false, b,
                                               (short)0, c, false, false);
  }
#pragma unroll
  for (int kb = 0; kb < K; kb += 32) {     // root features @ Wr^T
    v16h a = load_a_lds(a1 + kb);
    v16h b = load_b_f16(br + kb);
    c = __builtin_amdgcn_wmma_f32_16x16x32_f16(false, a, false, b,
                                               (short)0, c, false, false);
  }

  // C/D layout: VGPR r -> M = r + (lane>=16 ? 8 : 0), N = lane&15
  const float bv    = bias[n];
  const int   mbase = m0 + ((lane >> 4) << 3);
#pragma unroll
  for (int r = 0; r < 8; ++r) {
    float v = c[r] + bv;
    out[(size_t)(mbase + r) * FOUT + n] = v > 0.0f ? v : 0.0f;
  }
}

// ---------------------------------------------------------------------------
// Edge scatter: one thread per (edge, float4 chunk). Coalesced 512B gather
// per edge, global_atomic_add_f32 scatter into L2-resident agg buffers.
// ---------------------------------------------------------------------------
template <int LOGCH>
__global__ void scatter_add_kernel(const float* __restrict__ feat,
                                   const int* __restrict__ src,
                                   const int* __restrict__ dst,
                                   float* __restrict__ agg, int E) {
  const long long tid = (long long)blockIdx.x * blockDim.x + threadIdx.x;
  const int e = (int)(tid >> LOGCH);
  if (e >= E) return;
  const int c  = ((int)tid & ((1 << LOGCH) - 1)) << 2;
  const int ld = 4 << LOGCH;
  const int sN = src[e];
  const int dN = dst[e];
  const float4 v = *(const float4*)(feat + (size_t)sN * ld + c);
  float* ap = agg + (size_t)dN * ld + c;
  atomicAdd(ap + 0, v.x);
  atomicAdd(ap + 1, v.y);
  atomicAdd(ap + 2, v.z);
  atomicAdd(ap + 3, v.w);
}

__global__ void degree_kernel(const int* __restrict__ dst,
                              float* __restrict__ deg, int E) {
  const int e = blockIdx.x * blockDim.x + threadIdx.x;
  if (e < E) atomicAdd(&deg[dst[e]], 1.0f);
}

__global__ void invdeg_kernel(float* __restrict__ deg, int n) {
  const int i = blockIdx.x * blockDim.x + threadIdx.x;
  if (i < n) deg[i] = 1.0f / fmaxf(deg[i], 1.0f);
}

__global__ void zero_kernel(float* __restrict__ p, long long n) {
  const long long i = (long long)blockIdx.x * blockDim.x + threadIdx.x;
  if (i < n) p[i] = 0.0f;
}

__global__ void cvt_f16_kernel(const float* __restrict__ in,
                               _Float16* __restrict__ out, int n) {
  const int i = blockIdx.x * blockDim.x + threadIdx.x;
  if (i < n) out[i] = (_Float16)in[i];
}

__global__ void classify_kernel(const float* __restrict__ h2,
                                const float* __restrict__ Wc,
                                const float* __restrict__ bc,
                                float* __restrict__ out, int n) {
  const int i = blockIdx.x * blockDim.x + threadIdx.x;
  if (i >= n) return;
  const float4* p = (const float4*)(h2 + (size_t)i * D_HID2);
  const float4* w = (const float4*)Wc;
  float s = 0.0f;
#pragma unroll
  for (int j = 0; j < D_HID2 / 4; ++j) {
    const float4 a = p[j], b = w[j];
    s += a.x * b.x + a.y * b.y + a.z * b.z + a.w * b.w;
  }
  out[i] = s + bc[0];
}

// ---------------------------------------------------------------------------
extern "C" void kernel_launch(void* const* d_in, const int* in_sizes, int n_in,
                              void* d_out, int out_size, void* d_ws,
                              size_t ws_size, hipStream_t stream) {
  (void)n_in; (void)out_size; (void)ws_size;

  const float* x   = (const float*)d_in[0];
  const int*   ei  = (const int*)d_in[1];   // int32 (JAX x64 disabled)
  const float* W1l = (const float*)d_in[2];
  const float* W1r = (const float*)d_in[3];
  const float* b1  = (const float*)d_in[4];
  const float* W2l = (const float*)d_in[5];
  const float* W2r = (const float*)d_in[6];
  const float* b2  = (const float*)d_in[7];
  const float* Wc  = (const float*)d_in[8];
  const float* bc  = (const float*)d_in[9];

  const int N = in_sizes[0] / D_IN;   // 100000
  const int E = in_sizes[1] / 2;      // 1600000
  const int* src = ei;
  const int* dst = ei + E;

  // Workspace (floats): [bufA: N*128][h1: N*128][deg: N][f16 weights: 96KB]
  // bufA = agg1 for layer 1, then reused as agg2 (first N*64) + h2 (next N*64).
  float* ws   = (float*)d_ws;
  float* bufA = ws;
  float* h1   = ws + (size_t)N * D_HID;
  float* deg  = h1 + (size_t)N * D_HID;
  _Float16* wf16  = (_Float16*)(deg + N);   // 16B-aligned for N=100000
  _Float16* W1l16 = wf16;
  _Float16* W1r16 = W1l16 + D_HID * D_IN;
  _Float16* W2l16 = W1r16 + D_HID * D_IN;
  _Float16* W2r16 = W2l16 + D_HID2 * D_HID;

  float* agg1 = bufA;
  float* agg2 = bufA;
  float* h2   = bufA + (size_t)N * D_HID2;
  float* logits = (float*)d_out;

  const int B = 256;
  auto nblk = [](long long n, int b) { return (unsigned)((n + b - 1) / b); };

  // ---- one-time weight conversion to f16 (tiny) ----
  cvt_f16_kernel<<<nblk(D_HID * D_IN, B), B, 0, stream>>>(W1l, W1l16, D_HID * D_IN);
  cvt_f16_kernel<<<nblk(D_HID * D_IN, B), B, 0, stream>>>(W1r, W1r16, D_HID * D_IN);
  cvt_f16_kernel<<<nblk(D_HID2 * D_HID, B), B, 0, stream>>>(W2l, W2l16, D_HID2 * D_HID);
  cvt_f16_kernel<<<nblk(D_HID2 * D_HID, B), B, 0, stream>>>(W2r, W2r16, D_HID2 * D_HID);

  // ---- Layer 1 ----
  zero_kernel<<<nblk((long long)N * D_IN, B), B, 0, stream>>>(agg1, (long long)N * D_IN);
  zero_kernel<<<nblk(N, B), B, 0, stream>>>(deg, N);
  degree_kernel<<<nblk(E, B), B, 0, stream>>>(dst, deg, E);
  scatter_add_kernel<5><<<nblk((long long)E * 32, B), B, 0, stream>>>(x, src, dst, agg1, E);
  invdeg_kernel<<<nblk(N, B), B, 0, stream>>>(deg, N);
  sage_gemm_kernel<D_IN, D_HID>
      <<<N / 16, (D_HID / 16) * 32, 0, stream>>>(agg1, deg, x, W1l16, W1r16, b1, h1);

  // ---- Layer 2 (bufA reused) ----
  zero_kernel<<<nblk((long long)N * D_HID2, B), B, 0, stream>>>(agg2, (long long)N * D_HID2);
  scatter_add_kernel<4><<<nblk((long long)E * 16, B), B, 0, stream>>>(h1, src, dst, agg2, E);
  sage_gemm_kernel<D_HID, D_HID2>
      <<<N / 16, (D_HID2 / 16) * 32, 0, stream>>>(agg2, deg, h1, W2l16, W2r16, b2, h2);

  // ---- Classifier head ----
  classify_kernel<<<nblk(N, B), B, 0, stream>>>(h2, Wc, bc, logits, N);
}
